// FluxSingleStreamBlock_24627342475396
// MI455X (gfx1250) — compile-verified, matrix operand order checked by reading, not run on hardware
//
#include <hip/hip_runtime.h>
#include <hip/hip_bf16.h>
#include <math.h>
#include <stdint.h>

// ---------------------------------------------------------------------------
// Flux single-stream block for MI455X (gfx1250, wave32, WMMA + TDM).
// Heavy math: v_wmma_f32_16x16x32_bf16. Tile feeds: tensor_load_to_lds
// (Tensor Data Mover) with double-buffered LDS, TENSORcnt waits.
// ---------------------------------------------------------------------------

#define HID   3072
#define NH    24
#define HD    128
#define MLP   12288
#define RANK  32
#define LSEQ  4608
#define N1    (3*HID + MLP)   // 21504  (qkv + mlp)
#define K2    (HID + MLP)     // 15360  (attn + mlp_act)

typedef __bf16 bf16_t;
typedef __attribute__((ext_vector_type(8)))  __bf16 v8bf;
typedef __attribute__((ext_vector_type(16))) __bf16 v16bf;
typedef __attribute__((ext_vector_type(8)))  float  v8f;
typedef __attribute__((ext_vector_type(4)))  unsigned v4u;
typedef __attribute__((ext_vector_type(8)))  int      v8i;
typedef __attribute__((ext_vector_type(4)))  int      v4i;

#if defined(__has_builtin)
#  if __has_builtin(__builtin_amdgcn_tensor_load_to_lds) && __has_builtin(__builtin_amdgcn_s_wait_tensorcnt)
#    define HAS_TDM 1
#  endif
#endif
#ifndef HAS_TDM
#  define HAS_TDM 0
#endif

// Load a 16x32 bf16 A/B fragment from row-major [row][k] memory (ld elements
// per row). ISA 7.12.2 wave32 layout: lanes 0-15 -> K {0..7,16..23},
// lanes 16-31 -> K {8..15,24..31}; row = lane & 15.
__device__ __forceinline__ v16bf load_frag(const bf16_t* base, int ld) {
  const int lane = threadIdx.x & 31;
  const int row  = lane & 15;
  const int koff = (lane >> 4) << 3;       // 0 or 8
  const bf16_t* p = base + row * ld + koff;
  v16bf f;
  reinterpret_cast<v8bf*>(&f)[0] = *reinterpret_cast<const v8bf*>(p);
  reinterpret_cast<v8bf*>(&f)[1] = *reinterpret_cast<const v8bf*>(p + 16);
  return f;
}

__device__ __forceinline__ v8f wmma_bf16(v16bf a, v16bf b, v8f c) {
  return __builtin_amdgcn_wmma_f32_16x16x32_bf16(false, a, false, b,
                                                 (short)0, c, false, false);
}

#if HAS_TDM
// Issue one TDM load of a [tile_h x tile_w] bf16 tile (row-major, stride_e
// elements between rows) into LDS at byte offset lds_byte. pad_code selects
// the HW LDS padding interval: 4 -> pad every 32 DWORDs (128B rows, 64 elem),
// 5 -> every 64 DWORDs (256B rows, 128 elem). pad_amount = 4 DWORDs (16B),
// giving LDS row strides of 72 / 136 bf16 elements respectively.
__device__ __forceinline__ void tdm_issue(unsigned lds_byte, const void* gp,
                                          unsigned tile_w, unsigned tile_h,
                                          unsigned stride_e, unsigned pad_code) {
  unsigned long long ga = (unsigned long long)(uintptr_t)gp;
  v4u g0 = {};
  g0[0] = 1u;                                         // count=1 (valid), user D#
  g0[1] = lds_byte;                                   // lds_addr
  g0[2] = (unsigned)(ga & 0xffffffffu);               // global_addr[31:0]
  g0[3] = (unsigned)((ga >> 32) & 0x01ffffffu)        // global_addr[56:32]
        | (2u << 30);                                 // type = 2 ("image")
  v8i g1 = {};
  g1[0] = (int)((1u << 16)            // data_size = 1 -> 2 bytes
              | (1u << 20)            // pad_enable
              | (pad_code << 22)      // pad_interval
              | (3u << 25));          // pad_amount = 4 DWORDs (16B)
  g1[1] = 0;                          // atomic_barrier_addr | tensor_dim0 lo16
  g1[2] = 0x4000;                     // tensor_dim0 = 1<<30 (hi16) | dim1 lo16
  g1[3] = (int)(0x4000u | (tile_w << 16)); // tensor_dim1 = 1<<30 | tile_dim0
  g1[4] = (int)tile_h;                // tile_dim1 (tile_dim2 = 0)
  g1[5] = (int)stride_e;              // tensor_dim0_stride[31:0] (elements)
  g1[6] = 0;
  g1[7] = 0;
  v4i z4 = {};
#if defined(__clang_major__) && (__clang_major__ >= 23)
  v8i z8 = {};
  __builtin_amdgcn_tensor_load_to_lds(g0, g1, z4, z4, z8, 0);
#else
  __builtin_amdgcn_tensor_load_to_lds(g0, g1, z4, z4, 0);
#endif
}
#endif

__device__ __forceinline__ unsigned lds_off(const void* p) {
  return (unsigned)(uintptr_t)p;      // generic LDS ptr: addr[31:0] = LDS offset
}

// ------------------------------- tiny kernels ------------------------------

__global__ void k_silu(const float* __restrict__ vec, float* __restrict__ sv) {
  int i = blockIdx.x * 256 + threadIdx.x;
  float v = vec[i];
  sv[i] = v / (1.f + __expf(-v));
}

__global__ void k_mod_gemv(const float* __restrict__ sv,
                           const float* __restrict__ mw,
                           const float* __restrict__ mb,
                           float* __restrict__ mod) {
  int o = blockIdx.x * 256 + threadIdx.x;
  const float* row = mw + (size_t)o * HID;
  float acc = mb[o];
  for (int k = 0; k < HID; ++k) acc += sv[k] * row[k];
  mod[o] = acc;
}

// W_eff[n][k] = bf16( w[n][k] + sum_r lu[n][r] * ld[r][k] )
__global__ void k_weff(const float* __restrict__ w,
                       const float* __restrict__ lu,
                       const float* __restrict__ ld,
                       bf16_t* __restrict__ out, int Kd) {
  size_t idx = (size_t)blockIdx.x * 256 + threadIdx.x;
  size_t n = idx / (size_t)Kd;
  size_t k = idx % (size_t)Kd;
  float acc = w[idx];
  const float* lur = lu + n * RANK;
#pragma unroll
  for (int r = 0; r < RANK; ++r) acc += lur[r] * ld[(size_t)r * Kd + k];
  out[idx] = (bf16_t)acc;
}

__global__ void k_ln_mod(const float* __restrict__ x,
                         const float* __restrict__ mod,
                         bf16_t* __restrict__ xmod) {
  const int l = blockIdx.x, tid = threadIdx.x;
  __shared__ float red[256];
  const float* row = x + (size_t)l * HID;
  float s = 0.f;
  for (int c = tid; c < HID; c += 256) s += row[c];
  red[tid] = s; __syncthreads();
  for (int o = 128; o > 0; o >>= 1) { if (tid < o) red[tid] += red[tid + o]; __syncthreads(); }
  float mu = red[0] * (1.f / HID);
  __syncthreads();
  float v = 0.f;
  for (int c = tid; c < HID; c += 256) { float d = row[c] - mu; v += d * d; }
  red[tid] = v; __syncthreads();
  for (int o = 128; o > 0; o >>= 1) { if (tid < o) red[tid] += red[tid + o]; __syncthreads(); }
  float rstd = rsqrtf(red[0] * (1.f / HID) + 1e-6f);
  for (int c = tid; c < HID; c += 256) {
    float xn = (row[c] - mu) * rstd;
    xmod[(size_t)l * HID + c] = (bf16_t)(xn * (1.f + mod[HID + c]) + mod[c]);
  }
}

__global__ void k_qkv_prep(const bf16_t* __restrict__ qkvm,
                           const float* __restrict__ pe,
                           const float* __restrict__ rq,
                           const float* __restrict__ rk,
                           bf16_t* __restrict__ q_rot,
                           bf16_t* __restrict__ k_rot,
                           bf16_t* __restrict__ vt) {
  const int l = blockIdx.x, h = blockIdx.y, d = threadIdx.x;  // 128 threads
  const size_t rowb = (size_t)l * N1;
  float qv = (float)qkvm[rowb + h * HD + d];
  float kv = (float)qkvm[rowb + HID + h * HD + d];
  float vv = (float)qkvm[rowb + 2 * HID + h * HD + d];
  float sq = qv * qv, sk = kv * kv;
#pragma unroll
  for (int m = 1; m < 32; m <<= 1) { sq += __shfl_xor(sq, m, 32); sk += __shfl_xor(sk, m, 32); }
  __shared__ float pq[4], pk[4];
  if ((d & 31) == 0) { pq[d >> 5] = sq; pk[d >> 5] = sk; }
  __syncthreads();
  float mq = (pq[0] + pq[1] + pq[2] + pq[3]) * (1.f / HD);
  float mk = (pk[0] + pk[1] + pk[2] + pk[3]) * (1.f / HD);
  float qn = qv * rsqrtf(mq + 1e-6f) * rq[d];
  float kn = kv * rsqrtf(mk + 1e-6f) * rk[d];
  int p = d >> 1, i = d & 1;
  const float* rot = pe + ((size_t)l * 64 + p) * 4;
  float qpart = __shfl_xor(qn, 1, 32);
  float kpart = __shfl_xor(kn, 1, 32);
  float q0 = i ? qpart : qn, q1 = i ? qn : qpart;
  float k0 = i ? kpart : kn, k1 = i ? kn : kpart;
  float qo = rot[i * 2 + 0] * q0 + rot[i * 2 + 1] * q1;
  float ko = rot[i * 2 + 0] * k0 + rot[i * 2 + 1] * k1;
  size_t hl = ((size_t)h * LSEQ + l) * HD + d;
  q_rot[hl] = (bf16_t)qo;
  k_rot[hl] = (bf16_t)ko;
  vt[((size_t)h * HD + d) * LSEQ + l] = (bf16_t)vv;   // V transposed [h][d][l]
}

__global__ void k_gelu(const bf16_t* __restrict__ qkvm, bf16_t* __restrict__ cat) {
  size_t idx = (size_t)blockIdx.x * 256 + threadIdx.x;   // < LSEQ*MLP
  size_t l = idx / MLP, j = idx % MLP;
  float x = (float)qkvm[l * N1 + 3 * HID + j];
  float g = 0.5f * x * (1.f + tanhf(0.7978845608028654f * (x + 0.044715f * x * x * x)));
  cat[l * K2 + HID + j] = (bf16_t)g;
}

// --------------------------- WMMA tiled GEMM -------------------------------
// C[M,N] = A[M,K] * B[N,K]^T (+bias). Block = 256 thr (8 waves), tile 128x128,
// K-step 64, double-buffered LDS tiles filled by the Tensor Data Mover
// (fallback: cooperative b128 loads). LDS row stride = 72 elems (64 + 16B pad).
// EPI=0: store bf16.  EPI=1: dout = x + gate * (acc + bias) in f32.
template <int EPI>
__global__ void __launch_bounds__(256)
k_gemm(const bf16_t* __restrict__ A, const bf16_t* __restrict__ B,
       const float* __restrict__ bias, int M, int N, int K,
       bf16_t* __restrict__ outb, float* __restrict__ outf,
       const float* __restrict__ xres, const float* __restrict__ gate) {
  __shared__ bf16_t sA[2][128 * 72];
  __shared__ bf16_t sB[2][128 * 72];
  const int tid  = threadIdx.x;
  const int lane = tid & 31;
  const int wave = tid >> 5;
  const int wm = wave >> 1, wn = wave & 1;
  const int row0 = blockIdx.y * 128;
  const int col0 = blockIdx.x * 128;

  v8f acc[2][4] = {};

#if HAS_TDM
  if (wave == 0) {   // prologue: DMA first K tile into buffer 0
    tdm_issue(lds_off(&sA[0][0]), &A[(size_t)row0 * K], 64, 128, (unsigned)K, 4);
    tdm_issue(lds_off(&sB[0][0]), &B[(size_t)col0 * K], 64, 128, (unsigned)K, 4);
  }
#endif

  for (int kb = 0; kb < K; kb += 64) {
    const int buf = (kb >> 6) & 1;
    __syncthreads();                       // everyone done reading buf^1
#if HAS_TDM
    if (wave == 0) {
      if (kb + 64 < K) {                   // DMA next tile into the other buffer
        tdm_issue(lds_off(&sA[buf ^ 1][0]), &A[(size_t)row0 * K + kb + 64], 64, 128, (unsigned)K, 4);
        tdm_issue(lds_off(&sB[buf ^ 1][0]), &B[(size_t)col0 * K + kb + 64], 64, 128, (unsigned)K, 4);
        __builtin_amdgcn_s_wait_tensorcnt((short)2);  // current tile complete
      } else {
        __builtin_amdgcn_s_wait_tensorcnt((short)0);
      }
    }
#else
#pragma unroll
    for (int t = 0; t < 4; ++t) {          // 1024 16B-chunks per 128x64 tile
      int c = tid + t * 256;
      int r = c >> 3, kc = (c & 7) << 3;
      const bf16_t* ga = &A[(size_t)(row0 + r) * K + kb + kc];
      const bf16_t* gb = &B[(size_t)(col0 + r) * K + kb + kc];
      *reinterpret_cast<v8bf*>(&sA[buf][r * 72 + kc]) = *reinterpret_cast<const v8bf*>(ga);
      *reinterpret_cast<v8bf*>(&sB[buf][r * 72 + kc]) = *reinterpret_cast<const v8bf*>(gb);
      if (kb + 64 < K) {
        __builtin_prefetch(ga + 64, 0, 1);
        __builtin_prefetch(gb + 64, 0, 1);
      }
    }
#endif
    __syncthreads();                       // tile visible to all waves
#pragma unroll
    for (int kd = 0; kd < 2; ++kd) {
      v16bf af[2], bf_[4];
#pragma unroll
      for (int i = 0; i < 2; ++i)
        af[i] = load_frag(&sA[buf][(wm * 32 + i * 16) * 72 + kd * 32], 72);
#pragma unroll
      for (int j = 0; j < 4; ++j)
        bf_[j] = load_frag(&sB[buf][(wn * 64 + j * 16) * 72 + kd * 32], 72);
#pragma unroll
      for (int i = 0; i < 2; ++i)
#pragma unroll
        for (int j = 0; j < 4; ++j)
          acc[i][j] = wmma_bf16(af[i], bf_[j], acc[i][j]);
    }
  }

  const int nlo = lane & 15;
  const int mb_ = (lane >> 4) << 3;
#pragma unroll
  for (int i = 0; i < 2; ++i) {
#pragma unroll
    for (int j = 0; j < 4; ++j) {
      int cg = col0 + wn * 64 + j * 16 + nlo;
      float bv = bias[cg];
#pragma unroll
      for (int r = 0; r < 8; ++r) {
        int mg = row0 + wm * 32 + i * 16 + mb_ + r;
        float val = acc[i][j][r] + bv;
        if (EPI == 0) {
          outb[(size_t)mg * N + cg] = (bf16_t)val;
        } else {
          size_t o = (size_t)mg * N + cg;
          outf[o] = xres[o] + gate[cg] * val;
        }
      }
    }
  }
}

// ------------------------ flash attention (WMMA) ---------------------------
// Grid: (LSEQ/128, NH). Block 256 thr = 8 waves; wave owns 16 query rows.
// 64-key blocks double-buffered via TDM: K tile [64 x 128] (256B rows -> LDS
// stride 136), V^T tile [128 x 64] (128B rows -> stride 72). S = Q K^T via
// WMMA, online softmax with half-wave shfl reductions, P relaid through LDS,
// O += P V via WMMA.
__global__ void __launch_bounds__(256)
k_flash(const bf16_t* __restrict__ qr, const bf16_t* __restrict__ kr,
        const bf16_t* __restrict__ vt, bf16_t* __restrict__ cat) {
  __shared__ bf16_t sK[2][64 * 136];
  __shared__ bf16_t sV[2][128 * 72];
  __shared__ bf16_t sP[8 * 16 * 72];
  const int tid = threadIdx.x, lane = tid & 31, wave = tid >> 5;
  const int h  = blockIdx.y;
  const int q0 = blockIdx.x * 128 + wave * 16;
  const float scale = 0.08838834764831845f;   // 1/sqrt(128)
  const bf16_t* kh = kr + (size_t)h * LSEQ * HD;
  const bf16_t* vh = vt + (size_t)h * HD * LSEQ;

  v16bf qf[4];
#pragma unroll
  for (int kd = 0; kd < 4; ++kd)
    qf[kd] = load_frag(qr + ((size_t)h * LSEQ + q0) * HD + kd * 32, HD);

  v8f acc_o[8] = {};
  float rmax[8], rsum[8];
#pragma unroll
  for (int r = 0; r < 8; ++r) { rmax[r] = -1e30f; rsum[r] = 0.f; }

  bf16_t* sPw = sP + wave * 16 * 72;
  const int nlo = lane & 15;
  const int mb_ = (lane >> 4) << 3;

#if HAS_TDM
  if (wave == 0) {
    tdm_issue(lds_off(&sK[0][0]), kh, 128, 64, HD, 5);
    tdm_issue(lds_off(&sV[0][0]), vh, 64, 128, LSEQ, 4);
  }
#endif

  for (int kb = 0; kb < LSEQ; kb += 64) {
    const int buf = (kb >> 6) & 1;
    __syncthreads();
#if HAS_TDM
    if (wave == 0) {
      if (kb + 64 < LSEQ) {
        tdm_issue(lds_off(&sK[buf ^ 1][0]), kh + (size_t)(kb + 64) * HD, 128, 64, HD, 5);
        tdm_issue(lds_off(&sV[buf ^ 1][0]), vh + (kb + 64), 64, 128, LSEQ, 4);
        __builtin_amdgcn_s_wait_tensorcnt((short)2);
      } else {
        __builtin_amdgcn_s_wait_tensorcnt((short)0);
      }
    }
#else
#pragma unroll
    for (int t = 0; t < 4; ++t) {
      int c = tid + t * 256;
      { int r = c >> 4, kc = (c & 15) << 3;
        *reinterpret_cast<v8bf*>(&sK[buf][r * 136 + kc]) =
            *reinterpret_cast<const v8bf*>(&kh[(size_t)(kb + r) * HD + kc]); }
      { int r = c >> 3, jc = (c & 7) << 3;
        *reinterpret_cast<v8bf*>(&sV[buf][r * 72 + jc]) =
            *reinterpret_cast<const v8bf*>(&vh[(size_t)r * LSEQ + kb + jc]); }
    }
#endif
    __syncthreads();

    v8f s[4] = {};
#pragma unroll
    for (int j = 0; j < 4; ++j)
#pragma unroll
      for (int kd = 0; kd < 4; ++kd)
        s[j] = wmma_bf16(qf[kd], load_frag(&sK[buf][(j * 16) * 136 + kd * 32], 136), s[j]);

#pragma unroll
    for (int r = 0; r < 8; ++r) {
      float s0 = s[0][r] * scale, s1 = s[1][r] * scale;
      float s2 = s[2][r] * scale, s3 = s[3][r] * scale;
      float mx = fmaxf(fmaxf(s0, s1), fmaxf(s2, s3));
#pragma unroll
      for (int m = 1; m < 16; m <<= 1) mx = fmaxf(mx, __shfl_xor(mx, m, 32));
      float nm   = fmaxf(rmax[r], mx);
      float corr = __expf(rmax[r] - nm);
      rmax[r] = nm;
      float p0 = __expf(s0 - nm), p1 = __expf(s1 - nm);
      float p2 = __expf(s2 - nm), p3 = __expf(s3 - nm);
      float ls = p0 + p1 + p2 + p3;
#pragma unroll
      for (int m = 1; m < 16; m <<= 1) ls += __shfl_xor(ls, m, 32);
      rsum[r] = rsum[r] * corr + ls;
#pragma unroll
      for (int t = 0; t < 8; ++t) acc_o[t][r] *= corr;
      bf16_t* pr = sPw + (mb_ + r) * 72;
      pr[ 0 + nlo] = (bf16_t)p0;
      pr[16 + nlo] = (bf16_t)p1;
      pr[32 + nlo] = (bf16_t)p2;
      pr[48 + nlo] = (bf16_t)p3;
    }

#pragma unroll
    for (int ks = 0; ks < 2; ++ks) {
      v16bf pa = load_frag(sPw + ks * 32, 72);
#pragma unroll
      for (int t = 0; t < 8; ++t)
        acc_o[t] = wmma_bf16(pa, load_frag(&sV[buf][(t * 16) * 72 + ks * 32], 72), acc_o[t]);
    }
  }

#pragma unroll
  for (int t = 0; t < 8; ++t)
#pragma unroll
    for (int r = 0; r < 8; ++r) {
      float val = acc_o[t][r] / rsum[r];
      size_t m = (size_t)q0 + mb_ + r;
      cat[m * K2 + h * HD + t * 16 + nlo] = (bf16_t)val;
    }
}

// ------------------------------- launcher ----------------------------------

extern "C" void kernel_launch(void* const* d_in, const int* in_sizes, int n_in,
                              void* d_out, int out_size, void* d_ws, size_t ws_size,
                              hipStream_t stream) {
  (void)in_sizes; (void)n_in; (void)out_size; (void)ws_size;
  const float* x     = (const float*)d_in[0];
  const float* vec   = (const float*)d_in[1];
  const float* pe    = (const float*)d_in[2];
  const float* mod_w = (const float*)d_in[3];
  const float* mod_b = (const float*)d_in[4];
  const float* w1    = (const float*)d_in[5];
  const float* b1    = (const float*)d_in[6];
  const float* ld1   = (const float*)d_in[7];
  const float* lu1   = (const float*)d_in[8];
  const float* w2    = (const float*)d_in[9];
  const float* b2    = (const float*)d_in[10];
  const float* ld2   = (const float*)d_in[11];
  const float* lu2   = (const float*)d_in[12];
  const float* rq    = (const float*)d_in[13];
  const float* rk    = (const float*)d_in[14];

  char* ws = (char*)d_ws;
  size_t off = 0;
  auto carve = [&](size_t bytes) -> char* {
    char* p = ws + off;
    off = (off + bytes + 255) & ~(size_t)255;
    return p;
  };
  float*  sv    = (float*) carve((size_t)HID * 4);
  float*  mod   = (float*) carve((size_t)3 * HID * 4);
  bf16_t* xmod  = (bf16_t*)carve((size_t)LSEQ * HID * 2);
  bf16_t* w1e   = (bf16_t*)carve((size_t)N1 * HID * 2);
  bf16_t* w2e   = (bf16_t*)carve((size_t)HID * K2 * 2);
  bf16_t* qkvm  = (bf16_t*)carve((size_t)LSEQ * N1 * 2);
  bf16_t* q_rot = (bf16_t*)carve((size_t)NH * LSEQ * HD * 2);
  bf16_t* k_rot = (bf16_t*)carve((size_t)NH * LSEQ * HD * 2);
  bf16_t* vt    = (bf16_t*)carve((size_t)NH * LSEQ * HD * 2);
  bf16_t* cat   = (bf16_t*)carve((size_t)LSEQ * K2 * 2);

  k_silu<<<HID / 256, 256, 0, stream>>>(vec, sv);
  k_mod_gemv<<<(3 * HID) / 256, 256, 0, stream>>>(sv, mod_w, mod_b, mod);
  k_weff<<<(unsigned)(((size_t)N1 * HID) / 256), 256, 0, stream>>>(w1, lu1, ld1, w1e, HID);
  k_weff<<<(unsigned)(((size_t)HID * K2) / 256), 256, 0, stream>>>(w2, lu2, ld2, w2e, K2);
  k_ln_mod<<<LSEQ, 256, 0, stream>>>(x, mod, xmod);
  k_gemm<0><<<dim3(N1 / 128, LSEQ / 128), 256, 0, stream>>>(
      xmod, w1e, b1, LSEQ, N1, HID, qkvm, nullptr, nullptr, nullptr);
  k_qkv_prep<<<dim3(LSEQ, NH), 128, 0, stream>>>(qkvm, pe, rq, rk, q_rot, k_rot, vt);
  k_gelu<<<(unsigned)(((size_t)LSEQ * MLP) / 256), 256, 0, stream>>>(qkvm, cat);
  k_flash<<<dim3(LSEQ / 128, NH), 256, 0, stream>>>(q_rot, k_rot, vt, cat);
  k_gemm<1><<<dim3(HID / 128, LSEQ / 128), 256, 0, stream>>>(
      cat, w2e, b2, LSEQ, HID, K2, nullptr, (float*)d_out, x, mod + 2 * HID);
}
